// CombSubFastFac_66821101191256
// MI455X (gfx1250) — matrix operand order repaired and checked by reading, not compiled, not run on hardware
//
#include <hip/hip_runtime.h>
#include <hip/hip_bf16.h>
#include <math.h>

// ---------------------------------------------------------------------------
// CombSub vocoder for MI455X (gfx1250, wave32).
//  - GEMMs on v_wmma_f32_16x16x32_f16, double-buffered LDS tiles stored
//    pre-swizzled in WMMA fragment order (fragment = 2x ds_load_b128).
//    Next-tile global loads are register-staged and overlap the WMMAs.
//  - GEMM1 folds the 3 non-unit features into an exact rank-3 epilogue
//    correction; also emits an f16 copy of hidden for GEMM2's A tiles.
//  - Phase cumsum via closed-form frame prefix; STFT via packed complex FFT.
// Workspace: S (64 KB) + ctrl (~101 MB) + hidden_f16 (8.4 MB).
// ---------------------------------------------------------------------------

typedef __attribute__((ext_vector_type(16))) _Float16 v16h;
typedef __attribute__((ext_vector_type(8)))  _Float16 v8h;
typedef __attribute__((ext_vector_type(4)))  _Float16 v4h;
typedef __attribute__((ext_vector_type(8)))  float    v8f;

#define NB      8
#define TT      2048
#define HIDN    256
#define NCTRL   1539          // 3*(512+1)
#define NSAMP   (TT * 512)
#define MROWS   (NB * TT)     // 16384
#define PI_F    3.14159265358979323846f
#define SR_F    44100.0f
#define INV_SR  (1.0f / 44100.0f)

// ---------------------------------------------------------------------------
__global__ void zero_kernel(float4* p, int n4) {
    int i = blockIdx.x * blockDim.x + threadIdx.x;
    if (i < n4) p[i] = make_float4(0.f, 0.f, 0.f, 0.f);
}

// ---------------------------------------------------------------------------
// exclusive frame-level phase prefix S[b][t] = sum_{j<t} frameSum(j)/SR
__global__ void prefix_kernel(const float* __restrict__ f0,
                              float* __restrict__ S) {
    __shared__ float cs[256];
    const int b = blockIdx.x;
    const int tid = threadIdx.x;
    const float* fb = f0 + b * TT;
    float local[8];
    float sum = 0.0f;
#pragma unroll
    for (int i = 0; i < 8; ++i) {
        int t = tid * 8 + i;
        float a = fb[t];
        float c = (t < TT - 1) ? fb[t + 1] : a;
        float fs = (512.0f * a + (c - a) * 255.5f) * INV_SR;
        local[i] = sum;
        sum += fs;
    }
    cs[tid] = sum;
    __syncthreads();
    for (int off = 1; off < 256; off <<= 1) {
        float v = (tid >= off) ? cs[tid - off] : 0.0f;
        __syncthreads();
        cs[tid] += v;
        __syncthreads();
    }
    float base = cs[tid] - sum;
#pragma unroll
    for (int i = 0; i < 8; ++i)
        S[b * TT + tid * 8 + i] = base + local[i];
}

// ---------------------------------------------------------------------------
// hidden = tanh(units @ W_in[0:256] + rank3(f0,phase,vol) + b_in + spk_embed)
__global__ __launch_bounds__(128)
void gemm1_wmma(const float* __restrict__ units,
                const float* __restrict__ f0,
                const float* __restrict__ vol,
                const float* __restrict__ spk_embed,
                const float* __restrict__ W_in,
                const float* __restrict__ b_in,
                const int*   __restrict__ spk_id,
                const float* __restrict__ S,
                float* __restrict__ hidden,
                _Float16* __restrict__ hid16) {
    __shared__ _Float16 Asw[2][4][32][16];   // [buf][wave][lane][elem]
    __shared__ _Float16 Bsw[2][4][32][16];   // [buf][ntile][lane][elem]
    const int tid   = threadIdx.x;
    const int lane  = tid & 31;
    const int wave  = tid >> 5;
    const int mBase = blockIdx.x * 64;
    const int nBase = blockIdx.y * 64;

    // thread-constant staging decomposition
    const int aKk0   = (tid & 7) * 4;                  // A k-quad
    const int aRow   = tid >> 3;                       // 0..15
    const int aLane  = aRow | (((aKk0 >> 3) & 1) << 4);
    const int aI0    = (aKk0 & 7) + ((aKk0 & 16) ? 8 : 0);
    const int bN     = tid & 63;                       // B column in tile
    const int bKqB   = tid >> 6;                       // 0 or 1
    const int bNt    = bN >> 4;
    const int bCol   = bN & 15;

    float4 aReg[4];
    float  bReg[4][4];

    auto loadA = [&](int kt) {
        const int k0 = kt * 32;
#pragma unroll
        for (int i = 0; i < 4; ++i)
            aReg[i] = *(const float4*)
                &units[((size_t)(mBase + aRow + 16 * i) << 8) + k0 + aKk0];
    };
    auto loadB = [&](int kt) {
        const int k0 = kt * 32;
#pragma unroll
        for (int i = 0; i < 4; ++i) {
            int kk0 = (bKqB + 2 * i) * 4;
            const float* wp = &W_in[(size_t)(k0 + kk0) * HIDN + nBase + bN];
            bReg[i][0] = wp[0];
            bReg[i][1] = wp[HIDN];
            bReg[i][2] = wp[2 * HIDN];
            bReg[i][3] = wp[3 * HIDN];
        }
    };
    auto storeAB = [&](int buf) {
#pragma unroll
        for (int i = 0; i < 4; ++i) {
            v4h h; h.x = (_Float16)aReg[i].x; h.y = (_Float16)aReg[i].y;
            h.z = (_Float16)aReg[i].z; h.w = (_Float16)aReg[i].w;
            *(v4h*)&Asw[buf][i][aLane][aI0] = h;
        }
#pragma unroll
        for (int i = 0; i < 4; ++i) {
            int kk0 = (bKqB + 2 * i) * 4;
            v4h h; h.x = (_Float16)bReg[i][0]; h.y = (_Float16)bReg[i][1];
            h.z = (_Float16)bReg[i][2]; h.w = (_Float16)bReg[i][3];
            *(v4h*)&Bsw[buf][bNt][bCol | (((kk0 >> 3) & 1) << 4)][(kk0 & 7) + ((kk0 & 16) ? 8 : 0)] = h;
        }
    };

    v8f acc[4] = {};

    loadA(0); loadB(0); storeAB(0);
    __syncthreads();

    for (int kt = 0; kt < 8; ++kt) {
        const int cur = kt & 1;
        if (kt < 7) { loadA(kt + 1); loadB(kt + 1); }   // overlap with WMMA
        v16h af = *(const v16h*)&Asw[cur][wave][lane][0];
#pragma unroll
        for (int nt = 0; nt < 4; ++nt) {
            v16h bf = *(const v16h*)&Bsw[cur][nt][lane][0];
            acc[nt] = __builtin_amdgcn_wmma_f32_16x16x32_f16(
                false, af, false, bf, (short)0, acc[nt], false, false);
        }
        if (kt < 7) storeAB(cur ^ 1);
        __syncthreads();
    }

    // ---- epilogue: rank-3 correction + bias + speaker embed + tanh
    const int r0 = (lane < 16) ? 0 : 8;
    const int cc = lane & 15;
    const int sid = spk_id[mBase >> 11];   // 64-row tile never crosses batch
    const float* w_f0 = W_in + 256 * HIDN;
    const float* w_ph = W_in + 257 * HIDN;
    const float* w_vl = W_in + 258 * HIDN;
#pragma unroll
    for (int r = 0; r < 8; ++r) {
        int gr = mBase + (wave << 4) + r + r0;
        float fv = f0[gr];
        float x  = S[gr] + fv * INV_SR;    // inclusive cumsum @ frame start
        x -= rintf(x);
        float ph = 2.0f * PI_F * x;
        float vl = vol[gr];
#pragma unroll
        for (int nt = 0; nt < 4; ++nt) {
            int gc = nBase + nt * 16 + cc;
            float v = acc[nt][r]
                    + fv * w_f0[gc] + ph * w_ph[gc] + vl * w_vl[gc]
                    + b_in[gc] + spk_embed[sid * HIDN + gc];
            float h = tanhf(v);
            hidden[(size_t)gr * HIDN + gc] = h;
            hid16[(size_t)gr * HIDN + gc] = (_Float16)h;
        }
    }
}

// ---------------------------------------------------------------------------
// ctrl = hidden @ W_out + b_out   (M=16384, K=256, N=1539), A from f16 copy
__global__ __launch_bounds__(128)
void gemm2_wmma(const _Float16* __restrict__ hid16,
                const float* __restrict__ W_out,
                const float* __restrict__ b_out,
                float* __restrict__ ctrl) {
    __shared__ _Float16 Asw[2][4][32][16];
    __shared__ _Float16 Bsw[2][4][32][16];
    const int tid   = threadIdx.x;
    const int lane  = tid & 31;
    const int wave  = tid >> 5;
    const int mBase = blockIdx.x * 64;
    const int nBase = blockIdx.y * 64;

    // A: 64 rows x 4 k-octets (8 halves = 16B); 256 items / 128 thr = 2 each
    const int aKk0  = (tid & 3) * 8;                  // k-octet
    const int aRow  = tid >> 2;                       // 0..31
    const int aLane = (aRow & 15) | (((aKk0 >> 3) & 1) << 4);
    const int aI0   = (aKk0 & 16) ? 8 : 0;            // 8-elem span base
    const int bN    = tid & 63;
    const int bKqB  = tid >> 6;
    const int bNt   = bN >> 4;
    const int bCol  = bN & 15;
    const bool bOk  = (nBase + bN) < NCTRL;

    v8h   aReg[2];
    float bReg[4][4];

    auto loadA = [&](int kt) {
        const int k0 = kt * 32;
#pragma unroll
        for (int i = 0; i < 2; ++i)
            aReg[i] = *(const v8h*)
                &hid16[((size_t)(mBase + aRow + 32 * i) << 8) + k0 + aKk0];
    };
    auto loadB = [&](int kt) {
        const int k0 = kt * 32;
#pragma unroll
        for (int i = 0; i < 4; ++i) {
            int kk0 = (bKqB + 2 * i) * 4;
            if (bOk) {
                const float* wp = &W_out[(size_t)(k0 + kk0) * NCTRL + nBase + bN];
                bReg[i][0] = wp[0];
                bReg[i][1] = wp[NCTRL];
                bReg[i][2] = wp[2 * NCTRL];
                bReg[i][3] = wp[3 * NCTRL];
            } else {
                bReg[i][0] = bReg[i][1] = bReg[i][2] = bReg[i][3] = 0.0f;
            }
        }
    };
    auto storeAB = [&](int buf) {
#pragma unroll
        for (int i = 0; i < 2; ++i)
            *(v8h*)&Asw[buf][(aRow + 32 * i) >> 4][aLane][aI0] = aReg[i];
#pragma unroll
        for (int i = 0; i < 4; ++i) {
            int kk0 = (bKqB + 2 * i) * 4;
            v4h h; h.x = (_Float16)bReg[i][0]; h.y = (_Float16)bReg[i][1];
            h.z = (_Float16)bReg[i][2]; h.w = (_Float16)bReg[i][3];
            *(v4h*)&Bsw[buf][bNt][bCol | (((kk0 >> 3) & 1) << 4)][(kk0 & 7) + ((kk0 & 16) ? 8 : 0)] = h;
        }
    };

    v8f acc[4] = {};

    loadA(0); loadB(0); storeAB(0);
    __syncthreads();

    for (int kt = 0; kt < 8; ++kt) {
        const int cur = kt & 1;
        if (kt < 7) { loadA(kt + 1); loadB(kt + 1); }
        v16h af = *(const v16h*)&Asw[cur][wave][lane][0];
#pragma unroll
        for (int nt = 0; nt < 4; ++nt) {
            v16h bf = *(const v16h*)&Bsw[cur][nt][lane][0];
            acc[nt] = __builtin_amdgcn_wmma_f32_16x16x32_f16(
                false, af, false, bf, (short)0, acc[nt], false, false);
        }
        if (kt < 7) storeAB(cur ^ 1);
        __syncthreads();
    }

    const int r0 = (lane < 16) ? 0 : 8;
    const int cc = lane & 15;
#pragma unroll
    for (int nt = 0; nt < 4; ++nt) {
#pragma unroll
        for (int r = 0; r < 8; ++r) {
            int gr = mBase + (wave << 4) + r + r0;
            int gc = nBase + nt * 16 + cc;
            if (gc < NCTRL)
                ctrl[(size_t)gr * NCTRL + gc] = acc[nt][r] + b_out[gc];
        }
    }
}

// ---------------------------------------------------------------------------
// 1024-pt radix-2 DIT FFT in LDS; input must be bit-reverse permuted.
__device__ __forceinline__ void fft1024(float2* a, int tid) {
    for (int half = 1; half < 1024; half <<= 1) {
        float angf = -PI_F / (float)half;
        for (int j = tid; j < 512; j += 256) {
            int pos = j & (half - 1);
            int i0 = ((j & ~(half - 1)) << 1) | pos;
            int i1 = i0 + half;
            float wim, wre;
            __sincosf(angf * (float)pos, &wim, &wre);
            float2 u = a[i0], v = a[i1];
            float tre = wre * v.x - wim * v.y;
            float tim = wre * v.y + wim * v.x;
            a[i1].x = u.x - tre; a[i1].y = u.y - tim;
            a[i0].x = u.x + tre; a[i0].y = u.y + tim;
        }
        __syncthreads();
    }
}

// per-frame STFT filter + iSTFT + overlap-add.  Z = comb + i*noise.
__global__ __launch_bounds__(256)
void spectral_kernel(const float* __restrict__ f0,
                     const float* __restrict__ noise,
                     const float* __restrict__ ctrl,
                     const float* __restrict__ S,
                     float* __restrict__ sig_out) {
    __shared__ float2 Z[1024];
    const int f = blockIdx.x;        // frame 0..2048
    const int b = blockIdx.y;
    const int tid = threadIdx.x;

    for (int n = tid; n < 1024; n += 256) {
        int s = f * 512 + n - 512;
        float comb = 0.0f, noi = 0.0f;
        if (s >= 0 && s < NSAMP) {
            int t = s >> 9, k = s & 511;
            float f0a = f0[b * TT + t];
            float f0b = (t < TT - 1) ? f0[b * TT + t + 1] : f0a;
            float fr  = (float)k * (1.0f / 512.0f);
            float f0u = f0a + (f0b - f0a) * fr;
            float cum = (float)(k + 1) * f0a +
                        (f0b - f0a) * (1.0f / 512.0f) * (0.5f * (float)k * (float)(k + 1));
            float x = S[b * TT + t] + cum * INV_SR;
            x -= rintf(x);
            float z = SR_F * x / (f0u + 0.001f);
            float pz = PI_F * z;
            comb = (fabsf(z) < 1e-6f) ? 1.0f : __sinf(pz) / pz;
            noi  = noise[(size_t)b * NSAMP + s] * 2.0f - 1.0f;
        }
        float win = __sinf(PI_F * (float)n * (1.0f / 1024.0f)); // sqrt-Hann
        float2 v; v.x = comb * win; v.y = noi * win;
        Z[__brev((unsigned)n) >> 22] = v;
    }
    __syncthreads();
    fft1024(Z, tid);

    const int tf = (f < TT - 1) ? f : (TT - 1);      // filter frame (edge repeat)
    const float* crow = ctrl + (size_t)(b * TT + tf) * NCTRL;

    float2 yc_s[3], yy_s[3]; int ks[3]; int cnt = 0;
    for (int k = tid; k <= 512; k += 256) {
        int kn = (1024 - k) & 1023;
        float2 zk = Z[k], zn = Z[kn];
        float cre = 0.5f * (zk.x + zn.x);   // comb_fft
        float cim = 0.5f * (zk.y - zn.y);
        float nre = 0.5f * (zk.y + zn.y);   // noise_fft
        float nim = 0.5f * (zn.x - zk.x);
        float hm = crow[k], hp = crow[513 + k], nm = crow[1026 + k];
        float mag = __expf(hm);
        float sim, sre;
        __sincosf(PI_F * hp, &sim, &sre);
        sre *= mag; sim *= mag;
        float nf = __expf(nm) * (1.0f / 128.0f);
        float yre = cre * sre - cim * sim + nre * nf;
        float yim = cre * sim + cim * sre + nim * nf;
        float2 yc; yc.x = yre; yc.y = -yim;  // conj(Y[k])
        float2 yy; yy.x = yre; yy.y =  yim;
        yc_s[cnt] = yc; yy_s[cnt] = yy; ks[cnt] = k; ++cnt;
    }
    __syncthreads();
    for (int i = 0; i < cnt; ++i) {
        int k = ks[i], kn = (1024 - k) & 1023;
        Z[__brev((unsigned)k) >> 22] = yc_s[i];
        if (kn != k) Z[__brev((unsigned)kn) >> 22] = yy_s[i];
    }
    __syncthreads();
    fft1024(Z, tid);                 // FFT(conj(Y)) = 1024*conj(IFFT(Y))

    for (int n = tid; n < 1024; n += 256) {
        int s = f * 512 + n - 512;
        if (s >= 0 && s < NSAMP) {
            float win = __sinf(PI_F * (float)n * (1.0f / 1024.0f));
            float v = Z[n].x * (1.0f / 1024.0f) * win;
            atomicAdd(&sig_out[(size_t)b * NSAMP + s], v);
        }
    }
}

// ---------------------------------------------------------------------------
__global__ void copy_kernel(const float4* __restrict__ src,
                            float4* __restrict__ d1,
                            float4* __restrict__ d2, int n4) {
    int i = blockIdx.x * blockDim.x + threadIdx.x;
    if (i < n4) { float4 v = src[i]; d1[i] = v; d2[i] = v; }
}

// ---------------------------------------------------------------------------
extern "C" void kernel_launch(void* const* d_in, const int* in_sizes, int n_in,
                              void* d_out, int out_size, void* d_ws, size_t ws_size,
                              hipStream_t stream) {
    const float* units  = (const float*)d_in[0];
    const float* f0     = (const float*)d_in[1];
    const float* vol    = (const float*)d_in[2];
    const float* spk    = (const float*)d_in[3];
    const float* W_in   = (const float*)d_in[4];
    const float* b_in   = (const float*)d_in[5];
    const float* W_out  = (const float*)d_in[6];
    const float* b_out  = (const float*)d_in[7];
    const float* noise  = (const float*)d_in[8];
    const int*   spk_id = (const int*)d_in[9];

    float* out = (float*)d_out;
    const int sigN = NB * NSAMP;              // 8388608
    const int hidN = MROWS * HIDN;            // 4194304
    float* sig    = out;                      // output 0
    float* hidden = out + sigN;               // output 1
    float* sig2   = out + sigN + hidN;        // output 2 (copy)
    float* sig3   = out + 2 * sigN + hidN;    // output 3 (copy)

    float*    Spre  = (float*)d_ws;                                   // 64 KB
    float*    ctrl  = (float*)((char*)d_ws + 64 * 1024);              // ~101 MB
    _Float16* hid16 = (_Float16*)((char*)d_ws + 64 * 1024
                                  + (size_t)MROWS * NCTRL * 4);       // 8.4 MB

    zero_kernel<<<(sigN / 4 + 255) / 256, 256, 0, stream>>>((float4*)sig, sigN / 4);
    prefix_kernel<<<NB, 256, 0, stream>>>(f0, Spre);
    gemm1_wmma<<<dim3(MROWS / 64, HIDN / 64), 128, 0, stream>>>(
        units, f0, vol, spk, W_in, b_in, spk_id, Spre, hidden, hid16);
    gemm2_wmma<<<dim3(MROWS / 64, (NCTRL + 63) / 64), 128, 0, stream>>>(
        hid16, W_out, b_out, ctrl);
    spectral_kernel<<<dim3(TT + 1, NB), 256, 0, stream>>>(
        f0, noise, ctrl, Spre, sig);
    copy_kernel<<<(sigN / 4 + 255) / 256, 256, 0, stream>>>(
        (const float4*)sig, (float4*)sig2, (float4*)sig3, sigN / 4);
}
